// MultiHeadAttention_21208548508172
// MI455X (gfx1250) — matrix operand order repaired and checked by reading, not compiled, not run on hardware
//
#include <hip/hip_runtime.h>

// ---------------------------------------------------------------------------
// MHA forward for MI455X (gfx1250, wave32, WMMA, LDS-staged, TDM for V tiles)
//   d_out = [ output fp32 (B*S*D) | attn fp32 (B*H*S*S) ]
//   d_ws  = [ Qh f16 16MB | Kh f16 16MB | Vh f16 16MB | ctx f16 16MB ]
// ---------------------------------------------------------------------------

#define D_MODEL 1024
#define HEADS   16
#define D_K     64
#define BB      4
#define SS      2048
#define M_ROWS  (BB * SS)          // 8192

typedef __attribute__((ext_vector_type(16))) _Float16 v16h;
typedef __attribute__((ext_vector_type(8)))  _Float16 v8h;
typedef __attribute__((ext_vector_type(4)))  _Float16 v4h;
typedef __attribute__((ext_vector_type(8)))  float    v8f;

#if defined(__HIP_DEVICE_COMPILE__) && __has_builtin(__builtin_amdgcn_tensor_load_to_lds)
#define HAS_TDM 1
typedef __attribute__((ext_vector_type(4))) unsigned int u32x4;
typedef __attribute__((ext_vector_type(4))) int          i32x4;
typedef __attribute__((ext_vector_type(8))) int          i32x8;
#else
#define HAS_TDM 0
#endif

__device__ __forceinline__ v8f wmma_f16(v16h a, v16h b, v8f c) {
    return __builtin_amdgcn_wmma_f32_16x16x32_f16(false, a, false, b,
                                                  (short)0, c, false, false);
}

__device__ __forceinline__ v16h cat16(v8h lo, v8h hi) {
    return __builtin_shufflevector(lo, hi, 0,1,2,3,4,5,6,7,8,9,10,11,12,13,14,15);
}

// A fragment (16x32 f16): per-lane halves 0-15 at k=koff..koff+15,
// koff=(lane>>4)*16 (contiguous, matches B fragments). 16B-aligned row ptr.
__device__ __forceinline__ v16h a_frag_lds(const _Float16* row, int lane) {
    const int koff = (lane >> 4) << 4;
    v8h lo = *(const v8h*)(row + koff);
    v8h hi = *(const v8h*)(row + koff + 8);
    return cat16(lo, hi);
}
// B fragment (32x16 f16): per-lane k = h + (lane>>4)*16 -> 16 contiguous
// halves when LDS holds B^T rows ([n][k]); 16B-aligned rows (scores path).
__device__ __forceinline__ v16h b_frag_lds_nk(const _Float16* rowN, int lane) {
    const int koff = (lane >> 4) << 4;
    v8h lo = *(const v8h*)(rowN + koff);
    v8h hi = *(const v8h*)(rowN + koff + 8);
    return cat16(lo, hi);
}
// Same, but only 8B-aligned rows (padded B^T tiles): 4x ds_load_b64.
__device__ __forceinline__ v16h b_frag_lds_nk8(const _Float16* rowN, int lane) {
    const int koff = (lane >> 4) << 4;
    v4h c0 = *(const v4h*)(rowN + koff);
    v4h c1 = *(const v4h*)(rowN + koff + 4);
    v4h c2 = *(const v4h*)(rowN + koff + 8);
    v4h c3 = *(const v4h*)(rowN + koff + 12);
    return cat16(__builtin_shufflevector(c0, c1, 0,1,2,3,4,5,6,7),
                 __builtin_shufflevector(c2, c3, 0,1,2,3,4,5,6,7));
}
// B fragment gather from packed [k][LDN] layout (TDM-written tiles).
template <int LDN>
__device__ __forceinline__ v16h b_frag_lds_kn(const _Float16 (*lB)[LDN], int n, int lane) {
    const int kbase = (lane >> 4) << 4;
    v16h bf;
#pragma unroll
    for (int h = 0; h < 16; ++h) bf[h] = lB[kbase + h][n];
    return bf;
}

#if HAS_TDM
// TDM: load a 32x64 f16 tile (row-major, stride 64 elems) from global to LDS.
__device__ __forceinline__ void tdm_load_v_tile(const _Float16* gsrc, unsigned lds_off) {
    const unsigned long long ga = (unsigned long long)(size_t)gsrc;
    u32x4 g0;
    g0[0] = 1u;                                        // count=1
    g0[1] = lds_off;                                   // lds_addr
    g0[2] = (unsigned)(ga & 0xffffffffull);            // global_addr[31:0]
    g0[3] = (unsigned)((ga >> 32) & 0x1ffffffull) | (2u << 30);  // [56:32] | type=2
    i32x8 g1;
    g1[0] = (1 << 16);          // wg_mask=0 | data_size=1 (2B)
    g1[1] = (D_K << 16);        // abar=0 | tensor_dim0[15:0]=64
    g1[2] = (32 << 16);         // tensor_dim0[31:16]=0 | tensor_dim1[15:0]=32
    g1[3] = (D_K << 16);        // tensor_dim1[31:16]=0 | tile_dim0=64
    g1[4] = 32;                 // tile_dim1=32 | tile_dim2=0
    g1[5] = D_K;                // tensor_dim0_stride[31:0]=64
    g1[6] = 0;
    g1[7] = 0;
    const i32x4 gz = {0, 0, 0, 0};
#if defined(__clang_major__) && (__clang_major__ >= 23)
    const i32x8 gz8 = {0, 0, 0, 0, 0, 0, 0, 0};
    __builtin_amdgcn_tensor_load_to_lds(g0, g1, gz, gz, gz8, 0);
#else
    __builtin_amdgcn_tensor_load_to_lds(g0, g1, gz, gz, 0);
#endif
}
#endif

// ---------------------------------------------------------------------------
// Projection GEMM: out_f16[B,H,S,dk] = f16( X[8192,1024] @ W[1024,1024] + b )
// Block 256 thr = 8 waves; block tile 128x128; wave tile 32x64; K-step 32.
// W tile staged TRANSPOSED [n][k] so B fragments are pure ds_load_b64.
// Grid: 64 * 8 = 512 blocks.
// ---------------------------------------------------------------------------
__global__ __launch_bounds__(256) void mha_proj_gemm(const float* __restrict__ X,
                                                     const float* __restrict__ W,
                                                     const float* __restrict__ bias,
                                                     _Float16* __restrict__ out) {
    __shared__ _Float16 lA[128][40];    // 128x32 used; padded rows (20 dwords)
    __shared__ _Float16 lBt[128][36];   // B^T: [n][k], padded rows (18 dwords)
    const int t = threadIdx.x;
    const int lane = t & 31;
    const int w = t >> 5;
    const int wm = w & 3, wn = w >> 2;
    const int bm = blockIdx.x >> 3, bn = blockIdx.x & 7;
    const int m_base = bm * 128, n_base = bn * 128;

    v8f acc[2][4];
#pragma unroll
    for (int i = 0; i < 2; ++i)
#pragma unroll
        for (int j = 0; j < 4; ++j)
#pragma unroll
            for (int r = 0; r < 8; ++r) acc[i][j][r] = 0.0f;

    for (int k0 = 0; k0 < D_MODEL; k0 += 32) {
        __syncthreads();
        // stage A: 128x32 fp32 -> f16 (4 float4 per thread)
        for (int c = t; c < 1024; c += 256) {
            const int row = c >> 3, col4 = (c & 7) << 2;
            const float4 f = *(const float4*)&X[(size_t)(m_base + row) * D_MODEL + k0 + col4];
            v4h h4 = {(_Float16)f.x, (_Float16)f.y, (_Float16)f.z, (_Float16)f.w};
            *(v4h*)&lA[row][col4] = h4;
        }
        // stage B transposed: lanes sweep n (coalesced b32 loads), 4 k each
        for (int c = t; c < 1024; c += 256) {
            const int n = c & 127;
            const int kq = (c >> 7) << 2;              // 0,4,...,28
            const float* wp = &W[(size_t)(k0 + kq) * D_MODEL + n_base + n];
            v4h h4 = {(_Float16)wp[0], (_Float16)wp[D_MODEL],
                      (_Float16)wp[2 * D_MODEL], (_Float16)wp[3 * D_MODEL]};
            *(v4h*)&lBt[n][kq] = h4;
        }
        __syncthreads();
        const v16h a0 = a_frag_lds(&lA[wm * 32 + (lane & 15)][0], lane);
        const v16h a1 = a_frag_lds(&lA[wm * 32 + 16 + (lane & 15)][0], lane);
#pragma unroll
        for (int j = 0; j < 4; ++j) {
            const int n = wn * 64 + j * 16 + (lane & 15);
            const v16h bf = b_frag_lds_nk8(&lBt[n][0], lane);
            acc[0][j] = wmma_f16(a0, bf, acc[0][j]);
            acc[1][j] = wmma_f16(a1, bf, acc[1][j]);
        }
    }
    // store head-split: col n = head*64 + d ; row m = b*S + s
#pragma unroll
    for (int j = 0; j < 4; ++j) {
        const int ncol = n_base + wn * 64 + j * 16 + (lane & 15);
        const float bv = bias[ncol];
        const int hd = ncol >> 6, dd = ncol & 63;
#pragma unroll
        for (int tt = 0; tt < 2; ++tt)
#pragma unroll
            for (int r = 0; r < 8; ++r) {
                const int mg = m_base + wm * 32 + tt * 16 + r + ((lane >> 4) << 3);
                const int b = mg >> 11, s = mg & (SS - 1);
                out[(((size_t)(b * HEADS + hd) * SS) + s) * D_K + dd] =
                    (_Float16)(acc[tt][j][r] + bv);
            }
    }
}

// ---------------------------------------------------------------------------
// Scores: attn_raw[bh] = Qh[bh] @ Kh[bh]^T / 8.  Per bh: 2048x2048, K=64.
// Block tile 128x128, full K staged once; padded rows kill bank conflicts.
// Grid: 64 bh * 16 * 16 = 16384 blocks.
// ---------------------------------------------------------------------------
__global__ __launch_bounds__(256) void mha_scores(const _Float16* __restrict__ Qh,
                                                  const _Float16* __restrict__ Kh,
                                                  float* __restrict__ attn) {
    __shared__ _Float16 lQ[128][72];   // 64 used; 36-dword rows, 16B aligned
    __shared__ _Float16 lK[128][72];
    const int t = threadIdx.x;
    const int lane = t & 31;
    const int w = t >> 5;
    const int wm = w & 3, wn = w >> 2;
    const int bh = blockIdx.x >> 8;
    const int rem = blockIdx.x & 255;
    const int bm = rem >> 4, bn = rem & 15;
    const int m_base = bm * 128, n_base = bn * 128;
    const _Float16* Q = Qh + (size_t)bh * SS * D_K;
    const _Float16* K = Kh + (size_t)bh * SS * D_K;
    float* outp = attn + (size_t)bh * SS * SS;

    for (int c = t; c < 1024; c += 256) {   // 128 rows x 8 v8h chunks each
        const int row = c >> 3, col8 = (c & 7) << 3;
        *(v8h*)&lQ[row][col8] = *(const v8h*)&Q[(size_t)(m_base + row) * D_K + col8];
        *(v8h*)&lK[row][col8] = *(const v8h*)&K[(size_t)(n_base + row) * D_K + col8];
    }
    __syncthreads();

    v8f acc[2][4];
#pragma unroll
    for (int i = 0; i < 2; ++i)
#pragma unroll
        for (int j = 0; j < 4; ++j)
#pragma unroll
            for (int r = 0; r < 8; ++r) acc[i][j][r] = 0.0f;

#pragma unroll
    for (int k0 = 0; k0 < D_K; k0 += 32) {
        const v16h a0 = a_frag_lds(&lQ[wm * 32 + (lane & 15)][k0], lane);
        const v16h a1 = a_frag_lds(&lQ[wm * 32 + 16 + (lane & 15)][k0], lane);
#pragma unroll
        for (int j = 0; j < 4; ++j) {
            const int n = wn * 64 + j * 16 + (lane & 15);
            const v16h bf = b_frag_lds_nk(&lK[n][k0], lane);  // contiguous in k
            acc[0][j] = wmma_f16(a0, bf, acc[0][j]);
            acc[1][j] = wmma_f16(a1, bf, acc[1][j]);
        }
    }
    const float scale = 0.125f;  // 1/sqrt(64)
#pragma unroll
    for (int j = 0; j < 4; ++j) {
        const int ncol = n_base + wn * 64 + j * 16 + (lane & 15);
#pragma unroll
        for (int tt = 0; tt < 2; ++tt)
#pragma unroll
            for (int r = 0; r < 8; ++r) {
                const int mg = m_base + wm * 32 + tt * 16 + r + ((lane >> 4) << 3);
                outp[(size_t)mg * SS + ncol] = acc[tt][j][r] * scale;
            }
    }
}

// ---------------------------------------------------------------------------
// Softmax in place over rows of length S=2048; row in registers
// (8 floats/thread), wave32 shfl reductions + 8-entry LDS cross-wave step.
// ---------------------------------------------------------------------------
__global__ __launch_bounds__(256) void mha_softmax(float* __restrict__ attn) {
    __shared__ float wred[8];
    float* p = attn + (size_t)blockIdx.x * SS;
    const int t = threadIdx.x;
    const int lane = t & 31;
    const int w = t >> 5;

    float4 a = ((const float4*)p)[t * 2];
    float4 b = ((const float4*)p)[t * 2 + 1];

    float mx = fmaxf(fmaxf(fmaxf(a.x, a.y), fmaxf(a.z, a.w)),
                     fmaxf(fmaxf(b.x, b.y), fmaxf(b.z, b.w)));
#pragma unroll
    for (int off = 16; off > 0; off >>= 1) mx = fmaxf(mx, __shfl_xor(mx, off, 32));
    if (lane == 0) wred[w] = mx;
    __syncthreads();
    float bm = wred[0];
#pragma unroll
    for (int i = 1; i < 8; ++i) bm = fmaxf(bm, wred[i]);

    a.x = __expf(a.x - bm); a.y = __expf(a.y - bm);
    a.z = __expf(a.z - bm); a.w = __expf(a.w - bm);
    b.x = __expf(b.x - bm); b.y = __expf(b.y - bm);
    b.z = __expf(b.z - bm); b.w = __expf(b.w - bm);
    float sum = a.x + a.y + a.z + a.w + b.x + b.y + b.z + b.w;
#pragma unroll
    for (int off = 16; off > 0; off >>= 1) sum += __shfl_xor(sum, off, 32);
    __syncthreads();
    if (lane == 0) wred[w] = sum;
    __syncthreads();
    float bs = 0.0f;
#pragma unroll
    for (int i = 0; i < 8; ++i) bs += wred[i];
    const float inv = 1.0f / bs;

    a.x *= inv; a.y *= inv; a.z *= inv; a.w *= inv;
    b.x *= inv; b.y *= inv; b.z *= inv; b.w *= inv;
    ((float4*)p)[t * 2] = a;
    ((float4*)p)[t * 2 + 1] = b;
}

// ---------------------------------------------------------------------------
// Context: ctx[b,s,h,dk] = attn[bh] @ Vh[bh]   (M=2048, N=64, K=2048)
// Block tile 256x64; wave = 32 rows x 64 cols. V tile (32x64 f16) loaded via
// the Tensor Data Mover when available (wave 0 issues, s_wait_tensorcnt).
// lV stays packed [k][n]: it is the TDM destination (contiguous tile write).
// Grid: 64 bh * 8 = 512 blocks.
// ---------------------------------------------------------------------------
__global__ __launch_bounds__(256) void mha_ctx(const float* __restrict__ attn,
                                               const _Float16* __restrict__ Vh,
                                               _Float16* __restrict__ ctx) {
    __shared__ _Float16 lA[256][40];   // 256x32 used; padded rows
    __shared__ _Float16 lV[32][64];    // packed (TDM target)
    const int t = threadIdx.x;
    const int lane = t & 31;
    const int w = t >> 5;
    const int bh = blockIdx.x >> 3, bm = blockIdx.x & 7;
    const int m_base = bm * 256;
    const float*    A = attn + (size_t)bh * SS * SS;
    const _Float16* V = Vh + (size_t)bh * SS * D_K;

    v8f acc[2][4];
#pragma unroll
    for (int i = 0; i < 2; ++i)
#pragma unroll
        for (int j = 0; j < 4; ++j)
#pragma unroll
            for (int r = 0; r < 8; ++r) acc[i][j][r] = 0.0f;

#if HAS_TDM
    const unsigned lv_off = (unsigned)(size_t)&lV[0][0];  // flat LDS addr[31:0]
#endif

    for (int k0 = 0; k0 < SS; k0 += 32) {
        __syncthreads();
#if HAS_TDM
        if (w == 0) tdm_load_v_tile(V + (size_t)k0 * D_K, lv_off);
#else
        {   // manual V staging: 32x64 f16, one v8h per thread
            const int row = t >> 3, col8 = (t & 7) << 3;
            *(v8h*)&lV[row][col8] = *(const v8h*)&V[(size_t)(k0 + row) * D_K + col8];
        }
#endif
        // stage attn tile 256x32 fp32 -> f16 (8 float4 per thread)
        for (int c = t; c < 2048; c += 256) {
            const int row = c >> 3, col4 = (c & 7) << 2;
            const float4 f = *(const float4*)&A[(size_t)(m_base + row) * SS + k0 + col4];
            v4h h4 = {(_Float16)f.x, (_Float16)f.y, (_Float16)f.z, (_Float16)f.w};
            *(v4h*)&lA[row][col4] = h4;
        }
#if HAS_TDM
        if (w == 0) __builtin_amdgcn_s_wait_tensorcnt(0);
#endif
        __syncthreads();
        const v16h a0 = a_frag_lds(&lA[w * 32 + (lane & 15)][0], lane);
        const v16h a1 = a_frag_lds(&lA[w * 32 + 16 + (lane & 15)][0], lane);
#pragma unroll
        for (int j = 0; j < 4; ++j) {
            const int n = j * 16 + (lane & 15);
            const v16h bf = b_frag_lds_kn<64>(lV, n, lane);
            acc[0][j] = wmma_f16(a0, bf, acc[0][j]);
            acc[1][j] = wmma_f16(a1, bf, acc[1][j]);
        }
    }
    const int b = bh >> 4, hd = bh & 15;
#pragma unroll
    for (int j = 0; j < 4; ++j) {
        const int ncol = j * 16 + (lane & 15);
#pragma unroll
        for (int tt = 0; tt < 2; ++tt)
#pragma unroll
            for (int r = 0; r < 8; ++r) {
                const int s = m_base + w * 32 + tt * 16 + r + ((lane >> 4) << 3);
                ctx[(((size_t)(b * SS + s) * HEADS) + hd) * D_K + ncol] =
                    (_Float16)acc[tt][j][r];
            }
    }
}

// ---------------------------------------------------------------------------
// Output projection: out[8192,1024] = ctx_f16 @ Wo + bo (fp32 out).
// Same blocking as mha_proj_gemm; A staged from f16 source; Wo transposed.
// ---------------------------------------------------------------------------
__global__ __launch_bounds__(256) void mha_outproj(const _Float16* __restrict__ Xh,
                                                   const float* __restrict__ W,
                                                   const float* __restrict__ bias,
                                                   float* __restrict__ out) {
    __shared__ _Float16 lA[128][40];
    __shared__ _Float16 lBt[128][36];
    const int t = threadIdx.x;
    const int lane = t & 31;
    const int w = t >> 5;
    const int wm = w & 3, wn = w >> 2;
    const int bm = blockIdx.x >> 3, bn = blockIdx.x & 7;
    const int m_base = bm * 128, n_base = bn * 128;

    v8f acc[2][4];
#pragma unroll
    for (int i = 0; i < 2; ++i)
#pragma unroll
        for (int j = 0; j < 4; ++j)
#pragma unroll
            for (int r = 0; r < 8; ++r) acc[i][j][r] = 0.0f;

    for (int k0 = 0; k0 < D_MODEL; k0 += 32) {
        __syncthreads();
        // stage A (f16 source): 128x32 halves = 512 v8h chunks, 2 per thread
        for (int c = t; c < 512; c += 256) {
            const int row = c >> 2, col8 = (c & 3) << 3;
            *(v8h*)&lA[row][col8] =
                *(const v8h*)&Xh[(size_t)(m_base + row) * D_MODEL + k0 + col8];
        }
        // stage B transposed (coalesced along n)
        for (int c = t; c < 1024; c += 256) {
            const int n = c & 127;
            const int kq = (c >> 7) << 2;
            const float* wp = &W[(size_t)(k0 + kq) * D_MODEL + n_base + n];
            v4h h4 = {(_Float16)wp[0], (_Float16)wp[D_MODEL],
                      (_Float16)wp[2 * D_MODEL], (_Float16)wp[3 * D_MODEL]};
            *(v4h*)&lBt[n][kq] = h4;
        }
        __syncthreads();
        const v16h a0 = a_frag_lds(&lA[wm * 32 + (lane & 15)][0], lane);
        const v16h a1 = a_frag_lds(&lA[wm * 32 + 16 + (lane & 15)][0], lane);
#pragma unroll
        for (int j = 0; j < 4; ++j) {
            const int n = wn * 64 + j * 16 + (lane & 15);
            const v16h bf = b_frag_lds_nk8(&lBt[n][0], lane);
            acc[0][j] = wmma_f16(a0, bf, acc[0][j]);
            acc[1][j] = wmma_f16(a1, bf, acc[1][j]);
        }
    }
#pragma unroll
    for (int j = 0; j < 4; ++j) {
        const int ncol = n_base + wn * 64 + j * 16 + (lane & 15);
        const float bv = bias[ncol];
#pragma unroll
        for (int tt = 0; tt < 2; ++tt)
#pragma unroll
            for (int r = 0; r < 8; ++r) {
                const int mg = m_base + wm * 32 + tt * 16 + r + ((lane >> 4) << 3);
                out[(size_t)mg * D_MODEL + ncol] = acc[tt][j][r] + bv;
            }
    }
}

// ---------------------------------------------------------------------------
extern "C" void kernel_launch(void* const* d_in, const int* in_sizes, int n_in,
                              void* d_out, int out_size, void* d_ws, size_t ws_size,
                              hipStream_t stream) {
    const float* q  = (const float*)d_in[0];
    const float* k  = (const float*)d_in[1];
    const float* v  = (const float*)d_in[2];
    const float* Wq = (const float*)d_in[3];
    const float* bq = (const float*)d_in[4];
    const float* Wk = (const float*)d_in[5];
    const float* bk = (const float*)d_in[6];
    const float* Wv = (const float*)d_in[7];
    const float* bv = (const float*)d_in[8];
    const float* Wo = (const float*)d_in[9];
    const float* bo = (const float*)d_in[10];

    float* out  = (float*)d_out;                       // [B,S,D]
    float* attn = out + (size_t)M_ROWS * D_MODEL;      // [B,H,S,S]

    _Float16* qh  = (_Float16*)d_ws;                   // 16 MB each
    _Float16* kh  = qh + (size_t)M_ROWS * D_MODEL;
    _Float16* vh  = kh + (size_t)M_ROWS * D_MODEL;
    _Float16* ctx = vh + (size_t)M_ROWS * D_MODEL;

    const dim3 blk(256);

    // Q/K/V projections: 64 m-blocks x 8 n-blocks = 512 blocks each
    mha_proj_gemm<<<512, blk, 0, stream>>>(q, Wq, bq, qh);
    mha_proj_gemm<<<512, blk, 0, stream>>>(k, Wk, bk, kh);
    mha_proj_gemm<<<512, blk, 0, stream>>>(v, Wv, bv, vh);

    // scores: 64 bh * 256 tiles = 16384 blocks
    mha_scores<<<16384, blk, 0, stream>>>(qh, kh, attn);

    // softmax: one block per row
    mha_softmax<<<131072, blk, 0, stream>>>(attn);

    // ctx: 64 bh * 8 m-blocks = 512 blocks
    mha_ctx<<<512, blk, 0, stream>>>(attn, vh, ctx);

    // output projection
    mha_outproj<<<512, blk, 0, stream>>>(ctx, Wo, bo, out);
}
